// STGAT_30666066493970
// MI455X (gfx1250) — matrix operand order repaired and checked by reading, not compiled, not run on hardware
//
#include <hip/hip_runtime.h>

#define BB 128
#define WW 128
#define NN 64
#define HID 150
#define TT 128       // sequence length == W
#define GDIM 600     // 4*HID
#define GP   608     // GDIM padded to multiple of 32
#define HP   160     // HID padded to multiple of 32
#define DP   320     // 2*HID padded to multiple of 32

typedef __attribute__((ext_vector_type(16))) _Float16 v16h;
typedef __attribute__((ext_vector_type(8)))  float    v8f;

// ---------------------------------------------------------------------------
// WMMA GEMM:  C[M,N] = A[M,K] * B[K,N] (+bias) (+ReLU)
// PRECONDITION (guaranteed by caller via padding): M,N,K all multiples of 32,
// every row of A/B/C is 16-byte aligned.  No bounds checks -> the inner loop
// is 4 unconditional global_load_b128 (A) + 16 strided scalar loads (B) + f32->f16 packing + one v_wmma.
// 4 waves / 128-thread block, each wave owns a 16x16 C tile (block = 32x32).
// Batched via blockIdx.z with element strides (stride 0 = broadcast).
// Register layouts per CDNA5 ISA 7.12.2:
//   A  : lane(0-15|16-31) -> M row, elem i -> K = 8*half + (i&7) + 16*(i>>3)
//   B  : lane(0-15|16-31) -> N col, elem i -> K = 8*half + (i&7) + 16*(i>>3)
//   C/D: lane -> N = lane&15, M = elem + 8*half
// ---------------------------------------------------------------------------
template<int ACT>
__global__ void gemm_wmma(const float* __restrict__ A, const float* __restrict__ Bm,
                          float* __restrict__ C, const float* __restrict__ bias,
                          int K, int lda, int ldb, int ldc,
                          long long sA, long long sB, long long sC)
{
  const int lane = threadIdx.x & 31;
  const int wave = threadIdx.x >> 5;
  const int tn = (blockIdx.x * 2 + (wave & 1)) * 16;
  const int tm = (blockIdx.y * 2 + (wave >> 1)) * 16;
  const long long bz = blockIdx.z;
  A  += bz * sA;
  Bm += bz * sB;
  C  += bz * sC;

  const int half = lane >> 4;
  const float* Ar = A + (long long)(tm + (lane & 15)) * lda + 8 * half;
  v8f acc = {};

  for (int k0 = 0; k0 < K; k0 += 32) {
    const float* ap = Ar + k0;
    float4 a0 = *(const float4*)(ap);
    float4 a1 = *(const float4*)(ap + 4);
    float4 a2 = *(const float4*)(ap + 16);
    float4 a3 = *(const float4*)(ap + 20);
    const float* bp = Bm + (long long)(k0 + 8 * half) * ldb + tn + (lane & 15);
    const long long L = ldb;
    float4 b0 = make_float4(bp[0 * L], bp[1 * L], bp[2 * L], bp[3 * L]);
    float4 b1 = make_float4(bp[4 * L], bp[5 * L], bp[6 * L], bp[7 * L]);
    float4 b2 = make_float4(bp[16 * L], bp[17 * L], bp[18 * L], bp[19 * L]);
    float4 b3 = make_float4(bp[20 * L], bp[21 * L], bp[22 * L], bp[23 * L]);
    v16h a, b;
    a[0]  = (_Float16)a0.x; a[1]  = (_Float16)a0.y; a[2]  = (_Float16)a0.z; a[3]  = (_Float16)a0.w;
    a[4]  = (_Float16)a1.x; a[5]  = (_Float16)a1.y; a[6]  = (_Float16)a1.z; a[7]  = (_Float16)a1.w;
    a[8]  = (_Float16)a2.x; a[9]  = (_Float16)a2.y; a[10] = (_Float16)a2.z; a[11] = (_Float16)a2.w;
    a[12] = (_Float16)a3.x; a[13] = (_Float16)a3.y; a[14] = (_Float16)a3.z; a[15] = (_Float16)a3.w;
    b[0]  = (_Float16)b0.x; b[1]  = (_Float16)b0.y; b[2]  = (_Float16)b0.z; b[3]  = (_Float16)b0.w;
    b[4]  = (_Float16)b1.x; b[5]  = (_Float16)b1.y; b[6]  = (_Float16)b1.z; b[7]  = (_Float16)b1.w;
    b[8]  = (_Float16)b2.x; b[9]  = (_Float16)b2.y; b[10] = (_Float16)b2.z; b[11] = (_Float16)b2.w;
    b[12] = (_Float16)b3.x; b[13] = (_Float16)b3.y; b[14] = (_Float16)b3.z; b[15] = (_Float16)b3.w;
    acc = __builtin_amdgcn_wmma_f32_16x16x32_f16(false, a, false, b,
                                                 (short)0, acc, false, false);
  }

  const int cn = tn + (lane & 15);
  const float bv = bias ? bias[cn] : 0.f;
  float* Cp = C + (long long)(tm + 8 * half) * ldc + cn;
#pragma unroll
  for (int e = 0; e < 8; ++e) {
    float v = acc[e] + bv;
    if (ACT) v = v > 0.f ? v : 0.f;
    Cp[(long long)e * ldc] = v;
  }
}

static void gemm(hipStream_t st, const float* A, const float* B, float* C,
                 const float* bias, int M, int N, int K, int lda, int ldb, int ldc,
                 long long sA, long long sB, long long sC, int batch, int act)
{
  dim3 g((unsigned)(N / 32), (unsigned)(M / 32), (unsigned)batch);
  dim3 b(128, 1, 1);
  if (act) gemm_wmma<1><<<g, b, 0, st>>>(A, B, C, bias, K, lda, ldb, ldc, sA, sB, sC);
  else     gemm_wmma<0><<<g, b, 0, st>>>(A, B, C, bias, K, lda, ldb, ldc, sA, sB, sC);
}

// ---------------------------------------------------------------------------
// Weight packing: [O,I] row-major -> transposed, zero-padded [Kpad, Npad]
// (out[k][n] = w[n][k]).  Makes every GEMM B operand contiguous in N.
// ---------------------------------------------------------------------------
__global__ void pack_wT(const float* __restrict__ w, float* __restrict__ out,
                        int O, int I, int Npad, int Kpad)
{
  int idx = blockIdx.x * 256 + threadIdx.x;
  if (idx >= Npad * Kpad) return;
  int n = idx % Npad;
  int k = idx / Npad;
  out[idx] = (n < O && k < I) ? w[(long long)n * I + k] : 0.f;
}

// ---------------------------------------------------------------------------
// Elementwise / structural kernels
// ---------------------------------------------------------------------------
__global__ void copy_f32(const float* __restrict__ in, float* __restrict__ out, int n)
{ int i = blockIdx.x * 256 + threadIdx.x; if (i < n) out[i] = in[i]; }

__global__ void zero_f32(float* __restrict__ p, int n)
{ int i = blockIdx.x * 256 + threadIdx.x; if (i < n) p[i] = 0.f; }

// Conv1d over W (channels N, kernel KW, 'same' zero padding) + bias + ReLU.
template<int KW>
__global__ void conv_branch(const float* __restrict__ x, const float* __restrict__ w,
                            const float* __restrict__ bias, float* __restrict__ y)
{
  int gid = blockIdx.x * 256 + threadIdx.x;
  if (gid >= BB * WW * NN) return;
  int o  = gid & 63;
  int wo = (gid >> 6) & 127;
  int b  = gid >> 13;
  const int pad = (KW - 1) / 2;
  float acc = bias[o];
  for (int i = 0; i < NN; ++i) {
#pragma unroll
    for (int t = 0; t < KW; ++t) {
      int wp = wo + t - pad;
      if (wp >= 0 && wp < WW)
        acc += x[((long long)(b * WW + wp)) * NN + i] * w[(o * NN + i) * KW + t];
    }
  }
  y[gid] = acc > 0.f ? acc : 0.f;
}

// out[b][c][r] = in[b][r][c]  (in is [B,R,C])
__global__ void transpose_last2(const float* __restrict__ in, float* __restrict__ out,
                                int Bb, int R, int Cc)
{
  int idx = blockIdx.x * 256 + threadIdx.x;
  if (idx >= Bb * R * Cc) return;
  int c = idx % Cc;
  int r = (idx / Cc) % R;
  int b = idx / (R * Cc);
  out[(long long)b * R * Cc + (long long)c * R + r] = in[idx];
}

__global__ void rowdot2(const float* __restrict__ h, const float* __restrict__ asrc,
                        const float* __restrict__ adst, float* __restrict__ ssrc,
                        float* __restrict__ sdst, int rows, int cols)
{
  int r = blockIdx.x * 256 + threadIdx.x;
  if (r >= rows) return;
  float s1 = 0.f, s2 = 0.f;
  for (int w = 0; w < cols; ++w) {
    float v = h[(long long)r * cols + w];
    s1 += v * asrc[w];
    s2 += v * adst[w];
  }
  ssrc[r] = s1; sdst[r] = s2;
}

// dense GAT softmax (fully-connected graph + self loop): block=(b,j), 64 lanes=i
__global__ void gat_softmax(const float* __restrict__ ssrc, const float* __restrict__ sdst,
                            float* __restrict__ att)
{
  int bj = blockIdx.x;
  int b = bj >> 6, j = bj & 63, i = threadIdx.x;
  __shared__ float red[64];
  float e = ssrc[b * 64 + i] + sdst[b * 64 + j];
  e = e > 0.f ? e : 0.2f * e;      // LeakyReLU(0.2)
  red[i] = e; __syncthreads();
  for (int s = 32; s > 0; s >>= 1) { if (i < s) red[i] = fmaxf(red[i], red[i + s]); __syncthreads(); }
  float m = red[0]; __syncthreads();
  float ex = __expf(e - m);
  red[i] = ex; __syncthreads();
  for (int s = 32; s > 0; s >>= 1) { if (i < s) red[i] += red[i + s]; __syncthreads(); }
  att[(long long)bj * 64 + i] = ex / red[0];   // att[b][j][i] = alpha_{i->j}
}

// normalized triangular adjacency for the temporal GCN (constant across batch)
__global__ void build_mlow(float* __restrict__ M)
{
  int idx = blockIdx.x * 256 + threadIdx.x;
  if (idx >= WW * WW) return;
  int i = idx & 127, j = idx >> 7;   // M[j][i], deg[k]=k+1
  float v = 0.f;
  if (i < j)       v = rsqrtf((float)(i + 1)) * rsqrtf((float)(j + 1));
  else if (i == j) v = 1.0f / (float)(j + 1);
  M[idx] = v;
}

// faithful reshape-quirk residual:  hs[b,w,n] += g[b, n*W + w]
__global__ void residual_remap(float* __restrict__ hs, const float* __restrict__ g)
{
  int idx = blockIdx.x * 256 + threadIdx.x;
  if (idx >= BB * WW * NN) return;
  int n = idx & 63, w = (idx >> 6) & 127, b = idx >> 13;
  hs[idx] += g[(long long)b * (WW * NN) + n * WW + w];
}

__global__ void concat3(const float* __restrict__ a0, const float* __restrict__ a1,
                        const float* __restrict__ a2, float* __restrict__ out)
{
  int idx = blockIdx.x * 256 + threadIdx.x;
  if (idx >= BB * TT * 3 * NN) return;
  int c = idx % (3 * NN);
  int t = (idx / (3 * NN)) % TT;
  int b = idx / (3 * NN * TT);
  const float* s = (c < NN) ? a0 : ((c < 2 * NN) ? a1 : a2);
  out[idx] = s[((long long)(b * WW + t)) * NN + (c & 63)];
}

// rep[b,k,c] = h_end[b, (k*300+c)>>7]  (repeat_interleave quirk; W=128=1<<7)
// rep rows are DP=320 wide; pad columns written as zero.
__global__ void build_rep(const float* __restrict__ enc_seq, float* __restrict__ rep)
{
  int idx = blockIdx.x * 256 + threadIdx.x;
  if (idx >= BB * TT * DP) return;
  int c = idx % DP;
  int k = (idx / DP) % TT;
  int b = idx / (DP * TT);
  float v = 0.f;
  if (c < 2 * HID) {
    int p = k * (2 * HID) + c;
    v = enc_seq[((long long)(b * TT + (TT - 1))) * (2 * HID) + (p >> 7)];
  }
  rep[idx] = v;
}

// LSTM pointwise.  G rows GP wide, Xpre rows GP wide, H/C rows HP wide.
__global__ void lstm_point(const float* __restrict__ G, const float* __restrict__ Xpre,
                           const float* __restrict__ bih, const float* __restrict__ bhh,
                           float* __restrict__ H, float* __restrict__ Cst,
                           float* __restrict__ seq, int t, int outStride, int outOff)
{
  int idx = blockIdx.x * 256 + threadIdx.x;
  if (idx >= BB * HID) return;
  int b = idx / HID, j = idx % HID;
  const float* g  = G + (long long)b * GP;
  const float* xr = Xpre + ((long long)(b * TT + t)) * GP;
  float gi = g[j          ] + xr[j          ] + bih[j          ] + bhh[j          ];
  float gf = g[j +     HID] + xr[j +     HID] + bih[j +     HID] + bhh[j +     HID];
  float gg = g[j + 2 * HID] + xr[j + 2 * HID] + bih[j + 2 * HID] + bhh[j + 2 * HID];
  float go = g[j + 3 * HID] + xr[j + 3 * HID] + bih[j + 3 * HID] + bhh[j + 3 * HID];
  float si = 1.f / (1.f + __expf(-gi));
  float sf = 1.f / (1.f + __expf(-gf));
  float so = 1.f / (1.f + __expf(-go));
  float c = sf * Cst[b * HP + j] + si * tanhf(gg);
  float h = so * tanhf(c);
  Cst[b * HP + j] = c;
  H[b * HP + j] = h;
  seq[((long long)(b * TT + t)) * outStride + outOff + j] = h;
}

// ---------------------------------------------------------------------------
extern "C" void kernel_launch(void* const* d_in, const int* in_sizes, int n_in,
                              void* d_out, int out_size, void* d_ws, size_t ws_size,
                              hipStream_t stream)
{
  (void)in_sizes; (void)n_in; (void)out_size; (void)ws_size;
  const float* x        = (const float*)d_in[0];
  const float* conv2_w  = (const float*)d_in[3];
  const float* conv2_b  = (const float*)d_in[4];
  const float* conv3_w  = (const float*)d_in[5];
  const float* conv3_b  = (const float*)d_in[6];
  const float* gat_w    = (const float*)d_in[7];
  const float* gat_as   = (const float*)d_in[8];
  const float* gat_ad   = (const float*)d_in[9];
  const float* gat_bp   = (const float*)d_in[10];
  const float* gcn_w    = (const float*)d_in[11];
  const float* gcn_bp   = (const float*)d_in[12];
  const float* lstm_wih = (const float*)d_in[13];
  const float* lstm_whh = (const float*)d_in[14];
  const float* lstm_bih = (const float*)d_in[15];
  const float* lstm_bhh = (const float*)d_in[16];
  const float* dec_wih  = (const float*)d_in[17];
  const float* dec_whh  = (const float*)d_in[18];
  const float* dec_bih  = (const float*)d_in[19];
  const float* dec_bhh  = (const float*)d_in[20];
  const float* fc_w     = (const float*)d_in[21];
  const float* fc_b     = (const float*)d_in[22];
  float* out = (float*)d_out;

  // ---- workspace carve-out (floats; every block is a multiple of 4 floats) ----
  const long long E = (long long)BB * WW * NN;   // 1,048,576
  float* p = (float*)d_ws;
  float* hs0  = p; p += E;
  float* hs1  = p; p += E;
  float* hs2  = p; p += E;
  float* xp   = p; p += E;
  float* hgat = p; p += E;
  float* fbuf = p; p += E;
  float* zbuf = p; p += E;
  float* hz   = p; p += E;
  float* gbuf = p; p += E;
  float* ssrc = p; p += BB * NN;
  float* sdst = p; p += BB * NN;
  float* att  = p; p += (long long)BB * NN * NN;
  float* mlow = p; p += WW * WW;
  float* hcat = p; p += (long long)BB * TT * (3 * NN);
  float* xpre = p; p += (long long)BB * TT * GP;
  float* Gb   = p; p += BB * GP;
  float* Hb   = p; p += BB * HP;
  float* Cb   = p; p += BB * HP;
  float* encs = p; p += (long long)BB * TT * (2 * HID);
  float* rep  = p; p += (long long)BB * TT * DP;
  float* decc = p; p += (long long)BB * TT * DP;
  // packed (transposed + zero-padded) weights
  float* gatwT  = p; p += 6LL * WW * WW;          // 6 x [128,128]
  float* gcnwT  = p; p += 6LL * NN * NN;          // 6 x [64,64]
  float* wihT   = p; p += 2LL * (3 * NN) * GP;    // 2 x [192,608]
  float* whhT   = p; p += 2LL * HP * GP;          // 2 x [160,608]
  float* dwihT  = p; p += 2LL * DP * GP;          // 2 x [320,608]
  float* dwhhT  = p; p += 2LL * HP * GP;          // 2 x [160,608]
  float* fcT    = p; p += (long long)DP * NN;     // [320,64]

  auto ew = [](long long n) { return dim3((unsigned)((n + 255) / 256)); };

  // ---- pack all weights (transpose + pad) ----
  for (int tl = 0; tl < 6; ++tl) {
    pack_wT<<<ew(WW * WW), 256, 0, stream>>>(gat_w + (long long)tl * WW * WW,
                                             gatwT + (long long)tl * WW * WW,
                                             WW, WW, WW, WW);
    pack_wT<<<ew(NN * NN), 256, 0, stream>>>(gcn_w + (long long)tl * NN * NN,
                                             gcnwT + (long long)tl * NN * NN,
                                             NN, NN, NN, NN);
  }
  for (int d = 0; d < 2; ++d) {
    pack_wT<<<ew((long long)(3 * NN) * GP), 256, 0, stream>>>(
        lstm_wih + (long long)d * GDIM * (3 * NN), wihT + (long long)d * (3 * NN) * GP,
        GDIM, 3 * NN, GP, 3 * NN);
    pack_wT<<<ew((long long)HP * GP), 256, 0, stream>>>(
        lstm_whh + (long long)d * GDIM * HID, whhT + (long long)d * HP * GP,
        GDIM, HID, GP, HP);
    pack_wT<<<ew((long long)DP * GP), 256, 0, stream>>>(
        dec_wih + (long long)d * GDIM * (2 * HID), dwihT + (long long)d * DP * GP,
        GDIM, 2 * HID, GP, DP);
    pack_wT<<<ew((long long)HP * GP), 256, 0, stream>>>(
        dec_whh + (long long)d * GDIM * HID, dwhhT + (long long)d * HP * GP,
        GDIM, HID, GP, HP);
  }
  pack_wT<<<ew((long long)DP * NN), 256, 0, stream>>>(fc_w, fcT, NN, 2 * HID, NN, DP);

  // ---- input branches ----
  copy_f32<<<ew(E), 256, 0, stream>>>(x, hs0, (int)E);
  conv_branch<5><<<ew(E), 256, 0, stream>>>(x, conv2_w, conv2_b, hs1);
  conv_branch<7><<<ew(E), 256, 0, stream>>>(x, conv3_w, conv3_b, hs2);
  build_mlow<<<ew(WW * WW), 256, 0, stream>>>(mlow);

  // ---- 3 streams x 2 STGAT layers ----
  float* hsarr[3] = {hs0, hs1, hs2};
  for (int t = 0; t < 3; ++t) {
    for (int l = 0; l < 2; ++l) {
      int tl = t * 2 + l;
      const float* gwT = gatwT + (long long)tl * WW * WW;
      const float* gas = gat_as + tl * WW;
      const float* gad = gat_ad + tl * WW;
      const float* gb  = gat_bp + tl * WW;
      const float* cwT = gcnwT + (long long)tl * NN * NN;
      const float* cb  = gcn_bp + tl * NN;

      // GAT over N=64 fully-connected nodes, feature dim W=128
      transpose_last2<<<ew(E), 256, 0, stream>>>(hsarr[t], xp, BB, WW, NN);
      gemm(stream, xp, gwT, hgat, nullptr, BB * NN, WW, WW, WW, WW, WW,
           0, 0, 0, 1, 0);                                              // h = x·Wᵀ
      rowdot2<<<ew(BB * NN), 256, 0, stream>>>(hgat, gas, gad, ssrc, sdst, BB * NN, WW);
      gat_softmax<<<dim3(BB * NN), dim3(64), 0, stream>>>(ssrc, sdst, att);
      gemm(stream, att, hgat, fbuf, gb, NN, WW, NN, NN, WW, WW,
           (long long)NN * NN, (long long)NN * WW, (long long)NN * WW, BB, 1);

      // GCN over W=128 triangular temporal graph, feature dim N=64
      transpose_last2<<<ew(E), 256, 0, stream>>>(fbuf, zbuf, BB, NN, WW);
      gemm(stream, zbuf, cwT, hz, nullptr, BB * WW, NN, NN, NN, NN, NN,
           0, 0, 0, 1, 0);                                              // h = z·Wᵀ
      gemm(stream, mlow, hz, gbuf, cb, WW, NN, WW, WW, NN, NN,
           0, (long long)WW * NN, (long long)WW * NN, BB, 1);           // relu(Mlow·h+b)
      residual_remap<<<ew(E), 256, 0, stream>>>(hsarr[t], gbuf);
    }
  }

  concat3<<<ew((long long)BB * TT * 3 * NN), 256, 0, stream>>>(hs0, hs1, hs2, hcat);

  // ---- BiLSTM encoder ----
  for (int d = 0; d < 2; ++d) {
    gemm(stream, hcat, wihT + (long long)d * (3 * NN) * GP, xpre, nullptr,
         BB * TT, GP, 3 * NN, 3 * NN, GP, GP, 0, 0, 0, 1, 0);
    zero_f32<<<ew(BB * HP), 256, 0, stream>>>(Hb, BB * HP);
    zero_f32<<<ew(BB * HP), 256, 0, stream>>>(Cb, BB * HP);
    for (int s = 0; s < TT; ++s) {
      int tstep = (d == 0) ? s : (TT - 1 - s);
      gemm(stream, Hb, whhT + (long long)d * HP * GP, Gb, nullptr,
           BB, GP, HP, HP, GP, GP, 0, 0, 0, 1, 0);
      lstm_point<<<ew(BB * HID), 256, 0, stream>>>(Gb, xpre,
           lstm_bih + d * GDIM, lstm_bhh + d * GDIM, Hb, Cb, encs,
           tstep, 2 * HID, d * HID);
    }
  }

  build_rep<<<ew((long long)BB * TT * DP), 256, 0, stream>>>(encs, rep);

  // ---- BiLSTM decoder (writes concatenated halves into 320-wide rows) ----
  zero_f32<<<ew((long long)BB * TT * DP), 256, 0, stream>>>(decc, BB * TT * DP);
  for (int d = 0; d < 2; ++d) {
    gemm(stream, rep, dwihT + (long long)d * DP * GP, xpre, nullptr,
         BB * TT, GP, DP, DP, GP, GP, 0, 0, 0, 1, 0);
    zero_f32<<<ew(BB * HP), 256, 0, stream>>>(Hb, BB * HP);
    zero_f32<<<ew(BB * HP), 256, 0, stream>>>(Cb, BB * HP);
    for (int s = 0; s < TT; ++s) {
      int tstep = (d == 0) ? s : (TT - 1 - s);
      gemm(stream, Hb, dwhhT + (long long)d * HP * GP, Gb, nullptr,
           BB, GP, HP, HP, GP, GP, 0, 0, 0, 1, 0);
      lstm_point<<<ew(BB * HID), 256, 0, stream>>>(Gb, xpre,
           dec_bih + d * GDIM, dec_bhh + d * GDIM, Hb, Cb, decc,
           tstep, DP, d * HID);
    }
  }

  // ---- final FC: out = dec · fc_wᵀ + fc_b ----
  gemm(stream, decc, fcT, out, fc_b, BB * TT, NN, DP, DP, NN, NN,
       0, 0, 0, 1, 0);
}
